// MultiHeadAttention_5970004541617
// MI455X (gfx1250) — matrix operand order repaired and checked by reading, not compiled, not run on hardware
//
#include <hip/hip_runtime.h>
#include <hip/hip_bf16.h>

// ---------------------------------------------------------------------------
// MultiHeadAttention forward for MI455X (gfx1250), bf16 WMMA pipeline.
// B=4, T=2048, C=1024, H=16, d_k=64.
//
// Workspace layout (bytes), total 92,274,688 (~88 MB):
//   xb   @ 0         : x in bf16              (8192 x 1024)         16 MB
//   qb   @ 16 MB     : Q * (1/8) bf16, (bh, t, d) = (64,2048,64)    16 MB
//   kb   @ 32 MB     : K bf16, (bh, t, d)                           16 MB
//   vt   @ 48 MB     : V transposed bf16, (bh, d, t)                16 MB
//   ob   @ 64 MB     : attention output bf16 (8192 x 1024)          16 MB
//   wqt  @ 80 MB     : w_qkv^T bf16 (3072 x 1024)                    6 MB
//   wot  @ 86 MB     : w_out^T bf16 (1024 x 1024)                    2 MB
// ---------------------------------------------------------------------------

typedef __attribute__((ext_vector_type(16))) __bf16 v16bf;
typedef __attribute__((ext_vector_type(8)))  float  v8f;

#define WMMA_BF16(a, b, c) \
  __builtin_amdgcn_wmma_f32_16x16x32_bf16(false, (a), false, (b), (short)0, (c), false, false)

// A fragment (16x32 bf16, M x K) from row-major source, row stride ld (elems).
// Lane m (0-15): row m, K = {kh*8 .. kh*8+7} and {kh*8+16 .. kh*8+23}, kh = lane>>4.
static __device__ __forceinline__ v16bf load_a_frag(const __bf16* base, int ld, int lane) {
  const int m  = lane & 15;
  const int kh = lane >> 4;
  const __bf16* p = base + (size_t)m * ld + kh * 8;
  union { uint4 u[2]; v16bf v; } f;
  f.u[0] = *reinterpret_cast<const uint4*>(p);        // K = kh*8 + 0..7
  f.u[1] = *reinterpret_cast<const uint4*>(p + 16);   // K = kh*8 + 16..23
  return f.v;
}

// B fragment (32x16 bf16, K x N) from a source storing B^T row-major
// (column n of B == row n of source), row stride ld (elems).
// Lane n (0-15): column n, K = {kh*8 .. kh*8+7} and {kh*8+16 .. kh*8+23}, kh = lane>>4.
static __device__ __forceinline__ v16bf load_b_frag(const __bf16* baseT, int ld, int lane) {
  const int n  = lane & 15;
  const int kh = lane >> 4;
  const __bf16* p = baseT + (size_t)n * ld + kh * 8;
  union { uint4 u[2]; v16bf v; } f;
  f.u[0] = *reinterpret_cast<const uint4*>(p);        // K = kh*8 + 0..7
  f.u[1] = *reinterpret_cast<const uint4*>(p + 16);   // K = kh*8 + 16..23
  return f.v;
}

// ---------------------------------------------------------------------------
__global__ void __launch_bounds__(256)
f32_to_bf16_kernel(const float* __restrict__ src, __bf16* __restrict__ dst, int n) {
  for (int i = blockIdx.x * blockDim.x + threadIdx.x; i < n; i += gridDim.x * blockDim.x)
    dst[i] = (__bf16)src[i];
}

// dst[n*K + k] = (bf16) src[k*N + n]   (src is K x N row-major, dst is N x K)
__global__ void __launch_bounds__(256)
transpose_to_bf16_kernel(const float* __restrict__ src, __bf16* __restrict__ dst, int K, int N) {
  const int total = K * N;
  for (int i = blockIdx.x * blockDim.x + threadIdx.x; i < total; i += gridDim.x * blockDim.x) {
    const int n = i / K;
    const int k = i - n * K;
    dst[i] = (__bf16)src[(size_t)k * N + n];
  }
}

// ---------------------------------------------------------------------------
// QKV projection: (8192 x 1024) x (1024 x 3072). Each wave owns a 32x64 tile.
// Epilogue scatters into q (scaled), k, and transposed v.
__global__ void __launch_bounds__(256)
gemm_qkv_kernel(const __bf16* __restrict__ A, const __bf16* __restrict__ BT,
                __bf16* __restrict__ qb, __bf16* __restrict__ kb, __bf16* __restrict__ vt) {
  const int K = 1024;
  const int lane = threadIdx.x & 31;
  const int wid  = (blockIdx.x * blockDim.x + threadIdx.x) >> 5;
  const int wavesN = 3072 / 64;               // 48
  const int wm = wid / wavesN;
  const int wn = wid - wm * wavesN;
  const int row0 = wm * 32;
  const int col0 = wn * 64;

  v8f acc[2][4] = {};
  for (int k0 = 0; k0 < K; k0 += 32) {
    const v16bf a0 = load_a_frag(A + (size_t)row0 * K + k0, K, lane);
    const v16bf a1 = load_a_frag(A + (size_t)(row0 + 16) * K + k0, K, lane);
#pragma unroll
    for (int t = 0; t < 4; ++t) {
      const v16bf b = load_b_frag(BT + (size_t)(col0 + t * 16) * K + k0, K, lane);
      acc[0][t] = WMMA_BF16(a0, b, acc[0][t]);
      acc[1][t] = WMMA_BF16(a1, b, acc[1][t]);
    }
  }

  const int nl = lane & 15;
  const int mh = (lane >> 4) * 8;
#pragma unroll
  for (int mt = 0; mt < 2; ++mt)
#pragma unroll
    for (int t = 0; t < 4; ++t)
#pragma unroll
      for (int r = 0; r < 8; ++r) {
        const int row = row0 + mt * 16 + mh + r;
        const int col = col0 + t * 16 + nl;
        const float v = acc[mt][t][r];
        const int b     = row >> 11;           // row / 2048
        const int tt    = row & 2047;
        const int which = col >> 10;           // 0:q 1:k 2:v
        const int cc    = col & 1023;
        const int bh    = b * 16 + (cc >> 6);
        const int d     = cc & 63;
        if (which == 0)
          qb[((size_t)bh * 2048 + tt) * 64 + d] = (__bf16)(v * 0.125f);  // 1/sqrt(64)
        else if (which == 1)
          kb[((size_t)bh * 2048 + tt) * 64 + d] = (__bf16)v;
        else
          vt[((size_t)bh * 64 + d) * 2048 + tt] = (__bf16)v;
      }
}

// ---------------------------------------------------------------------------
// Causal flash attention. One wave per 16-row query tile per (b,h).
// grid = 2048 blocks x 128 threads (4 waves/block).
__global__ void __launch_bounds__(128)
attn_kernel(const __bf16* __restrict__ qb, const __bf16* __restrict__ kb,
            const __bf16* __restrict__ vt, __bf16* __restrict__ ob) {
  __shared__ __align__(16) __bf16 pbuf_all[4][16 * 32];
  const int lane  = threadIdx.x & 31;
  const int w     = threadIdx.x >> 5;
  const int wid   = blockIdx.x * 4 + w;        // 0..8191
  const int qtile = wid & 127;
  const int bh    = wid >> 7;                  // 0..63
  const int qbase = qtile << 4;
  const int nl = lane & 15;
  const int kh = lane >> 4;
  const int mh = kh << 3;

  const __bf16* Q  = qb + (size_t)bh * (2048 * 64);
  const __bf16* Kt = kb + (size_t)bh * (2048 * 64);
  const __bf16* Vt = vt + (size_t)bh * (64 * 2048);
  __bf16* pb = pbuf_all[w];

  const v16bf qf0 = load_a_frag(Q + (size_t)qbase * 64, 64, lane);      // d = 0..31
  const v16bf qf1 = load_a_frag(Q + (size_t)qbase * 64 + 32, 64, lane); // d = 32..63

  v8f o0 = {}, o1 = {}, o2 = {}, o3 = {};
  float m_run[8], l_run[8];
#pragma unroll
  for (int r = 0; r < 8; ++r) { m_run[r] = -1e30f; l_run[r] = 0.0f; }

  const int kend = qbase + 16;                 // exclusive: keys 0 .. qbase+15
  for (int ks = 0; ks < kend; ks += 32) {
    __builtin_prefetch(Kt + (size_t)(ks + 32) * 64, 0, 0);
    // ---- S = (Q/8) @ K^T : two 16x16 tiles, K-dim = d = 64 (2 WMMAs each)
    v8f s0, s1;
    {
      const v16bf b0 = load_b_frag(Kt + (size_t)ks * 64, 64, lane);
      const v16bf b1 = load_b_frag(Kt + (size_t)ks * 64 + 32, 64, lane);
      v8f z = {};
      z  = WMMA_BF16(qf0, b0, z);
      s0 = WMMA_BF16(qf1, b1, z);
    }
    {
      const v16bf b0 = load_b_frag(Kt + (size_t)(ks + 16) * 64, 64, lane);
      const v16bf b1 = load_b_frag(Kt + (size_t)(ks + 16) * 64 + 32, 64, lane);
      v8f z = {};
      z  = WMMA_BF16(qf0, b0, z);
      s1 = WMMA_BF16(qf1, b1, z);
    }
    // ---- causal mask + online softmax. Row m = mh + r lives at accumulator
    //      index r across the 16 lanes sharing kh -> shfl_xor reductions.
    float p0[8], p1[8];
#pragma unroll
    for (int r = 0; r < 8; ++r) {
      const int qrow = qbase + mh + r;
      float v0 = (ks + nl      <= qrow) ? s0[r] : -1e30f;
      float v1 = (ks + 16 + nl <= qrow) ? s1[r] : -1e30f;
      float bm = fmaxf(v0, v1);
#pragma unroll
      for (int off = 1; off < 16; off <<= 1)
        bm = fmaxf(bm, __shfl_xor(bm, off, 32));
      const float mn   = fmaxf(m_run[r], bm);
      const float corr = __expf(m_run[r] - mn);
      m_run[r] = mn;
      p0[r] = __expf(v0 - mn);
      p1[r] = __expf(v1 - mn);
      float rs = p0[r] + p1[r];
#pragma unroll
      for (int off = 1; off < 16; off <<= 1)
        rs += __shfl_xor(rs, off, 32);
      l_run[r] = l_run[r] * corr + rs;
      o0[r] *= corr; o1[r] *= corr; o2[r] *= corr; o3[r] *= corr;
    }
    // ---- P (16x32) through per-wave LDS to convert C-layout -> A-layout.
#pragma unroll
    for (int r = 0; r < 8; ++r) {
      pb[(mh + r) * 32 + nl]      = (__bf16)p0[r];
      pb[(mh + r) * 32 + 16 + nl] = (__bf16)p1[r];
    }
    asm volatile("s_wait_dscnt 0" ::: "memory");   // DS in-order per wave; RAW safe
    const v16bf pa = load_a_frag(pb, 32, lane);
    // ---- O += P @ V : V^T rows are d, contiguous over keys.
    {
      const v16bf bv = load_b_frag(Vt + (size_t)(0 * 16) * 2048 + ks, 2048, lane);
      o0 = WMMA_BF16(pa, bv, o0);
    }
    {
      const v16bf bv = load_b_frag(Vt + (size_t)(1 * 16) * 2048 + ks, 2048, lane);
      o1 = WMMA_BF16(pa, bv, o1);
    }
    {
      const v16bf bv = load_b_frag(Vt + (size_t)(2 * 16) * 2048 + ks, 2048, lane);
      o2 = WMMA_BF16(pa, bv, o2);
    }
    {
      const v16bf bv = load_b_frag(Vt + (size_t)(3 * 16) * 2048 + ks, 2048, lane);
      o3 = WMMA_BF16(pa, bv, o3);
    }
  }

  // ---- normalize and store to (B*T, C) row-major bf16
  const int b = bh >> 4, h = bh & 15;
#pragma unroll
  for (int r = 0; r < 8; ++r) {
    const float inv = 1.0f / l_run[r];
    const size_t row = (size_t)b * 2048 + qbase + mh + r;
    const size_t base = row * 1024 + h * 64 + nl;
    ob[base +  0] = (__bf16)(o0[r] * inv);
    ob[base + 16] = (__bf16)(o1[r] * inv);
    ob[base + 32] = (__bf16)(o2[r] * inv);
    ob[base + 48] = (__bf16)(o3[r] * inv);
  }
}

// ---------------------------------------------------------------------------
// Output projection: (8192 x 1024) x (1024 x 1024) -> f32 out.
__global__ void __launch_bounds__(256)
gemm_out_kernel(const __bf16* __restrict__ A, const __bf16* __restrict__ BT,
                float* __restrict__ out) {
  const int K = 1024;
  const int lane = threadIdx.x & 31;
  const int wid  = (blockIdx.x * blockDim.x + threadIdx.x) >> 5;
  const int wavesN = 1024 / 64;               // 16
  const int wm = wid / wavesN;
  const int wn = wid - wm * wavesN;
  const int row0 = wm * 32;
  const int col0 = wn * 64;

  v8f acc[2][4] = {};
  for (int k0 = 0; k0 < K; k0 += 32) {
    const v16bf a0 = load_a_frag(A + (size_t)row0 * K + k0, K, lane);
    const v16bf a1 = load_a_frag(A + (size_t)(row0 + 16) * K + k0, K, lane);
#pragma unroll
    for (int t = 0; t < 4; ++t) {
      const v16bf b = load_b_frag(BT + (size_t)(col0 + t * 16) * K + k0, K, lane);
      acc[0][t] = WMMA_BF16(a0, b, acc[0][t]);
      acc[1][t] = WMMA_BF16(a1, b, acc[1][t]);
    }
  }
  const int nl = lane & 15;
  const int mh = (lane >> 4) * 8;
#pragma unroll
  for (int mt = 0; mt < 2; ++mt)
#pragma unroll
    for (int t = 0; t < 4; ++t)
#pragma unroll
      for (int r = 0; r < 8; ++r) {
        const size_t row = (size_t)(row0 + mt * 16 + mh + r);
        const int col = col0 + t * 16 + nl;
        out[row * 1024 + col] = acc[mt][t][r];
      }
}

// ---------------------------------------------------------------------------
extern "C" void kernel_launch(void* const* d_in, const int* in_sizes, int n_in,
                              void* d_out, int out_size, void* d_ws, size_t ws_size,
                              hipStream_t stream) {
  (void)in_sizes; (void)n_in; (void)out_size;
  const float* x     = (const float*)d_in[0];   // (4,2048,1024)
  const float* w_qkv = (const float*)d_in[1];   // (1024,3072)
  const float* w_out = (const float*)d_in[2];   // (1024,1024)
  float* out = (float*)d_out;                   // (4,2048,1024)

  char* ws = (char*)d_ws;                       // requires >= 92,274,688 bytes
  __bf16* xb  = (__bf16*)(ws + 0);
  __bf16* qb  = (__bf16*)(ws + 16777216);
  __bf16* kb  = (__bf16*)(ws + 33554432);
  __bf16* vt  = (__bf16*)(ws + 50331648);
  __bf16* ob  = (__bf16*)(ws + 67108864);
  __bf16* wqt = (__bf16*)(ws + 83886080);
  __bf16* wot = (__bf16*)(ws + 90177536);

  f32_to_bf16_kernel<<<4096, 256, 0, stream>>>(x, xb, 8192 * 1024);
  transpose_to_bf16_kernel<<<2048, 256, 0, stream>>>(w_qkv, wqt, 1024, 3072);
  transpose_to_bf16_kernel<<<1024, 256, 0, stream>>>(w_out, wot, 1024, 1024);

  // 256 m-tiles * 48 n-tiles = 12288 waves, 8 waves/block -> 1536 blocks
  gemm_qkv_kernel<<<1536, 256, 0, stream>>>(xb, wqt, qb, kb, vt);

  // 64 (b,h) * 128 q-tiles = 8192 waves, 4 waves/block -> 2048 blocks
  attn_kernel<<<2048, 128, 0, stream>>>(qb, kb, vt, ob);

  // 256 m-tiles * 16 n-tiles = 4096 waves -> 512 blocks
  gemm_out_kernel<<<512, 256, 0, stream>>>(ob, wot, out);
}
